// RSSM_45363444580940
// MI455X (gfx1250) — compile-verified, matrix operand order checked by reading, not run on hardware
//
#include <hip/hip_runtime.h>

// ---------------------------------------------------------------------------
// RSSM for MI455X / gfx1250 : bf16 WMMA (f32 accum), async-to-LDS staging,
// persistent-scan kernel with device-wide barrier, fused 4-layer decoder.
// ---------------------------------------------------------------------------

#define DEVINL __device__ __forceinline__

typedef __attribute__((ext_vector_type(16))) __bf16 v16bf;
typedef __attribute__((ext_vector_type(8)))  float   v8f;
typedef __attribute__((ext_vector_type(8)))  __bf16  bf16x8;

static constexpr int Bb   = 128;
static constexpr int Tt   = 256;
static constexpr int DIN  = 1024;
static constexpr int DDET = 1024;
static constexpr int DSTO = 256;
static constexpr int DCAT = DDET + DSTO;   // 1280 decoder input width
static constexpr int ROWS = Bb * Tt;       // 32768
static constexpr int SCAN_BLOCKS = 64;     // persistent scan grid (8 waves each)
static constexpr int DRT  = 32;            // decoder row tile

// ---- WMMA helpers ---------------------------------------------------------
// A-fragment (16x32 bf16, row-major, ld in elements). ISA 7.12.2 layout:
// lane L (L<16): row M=L, elems 0..7 = K 0..7, elems 8..15 = K 16..23
// lane L+16   : row M=L, elems 0..7 = K 8..15, elems 8..15 = K 24..31
// Weights stored [N,K] load with the same per-lane pattern as B-fragments.
DEVINL v16bf load_frag_nk(const __bf16* p, int ld) {
  const int lane = threadIdx.x & 31;
  const __bf16* r = p + (long)(lane & 15) * ld + ((lane >> 4) << 3);
  bf16x8 lo = *(const bf16x8*)(r);
  bf16x8 hi = *(const bf16x8*)(r + 16);
  v16bf f;
#pragma unroll
  for (int i = 0; i < 8; ++i) { f[i] = lo[i]; f[i + 8] = hi[i]; }
  return f;
}

DEVINL v8f wmma_bf16(v16bf a, v16bf b, v8f c) {
  return __builtin_amdgcn_wmma_f32_16x16x32_bf16(false, a, false, b,
                                                 (short)0, c, false, false);
}

// C/D tile mapping: element v of lane l -> row = 8*(l>>4)+v, col = l&15
DEVINL int dcol()  { return threadIdx.x & 15; }
DEVINL int drow0() { return ((threadIdx.x & 31) >> 4) << 3; }

DEVINL float sigmoidf_(float x) { return 1.0f / (1.0f + __expf(-x)); }

// ---- device-wide split barrier for the persistent scan kernel -------------
DEVINL void grid_sync(unsigned* cnt, unsigned target) {
  __threadfence();                                  // release all our stores
  __syncthreads();
  if (threadIdx.x == 0) {
    __hip_atomic_fetch_add(cnt, 1u, __ATOMIC_RELEASE, __HIP_MEMORY_SCOPE_AGENT);
    while (__hip_atomic_load(cnt, __ATOMIC_RELAXED, __HIP_MEMORY_SCOPE_AGENT) < target)
      __builtin_amdgcn_s_sleep(1);
  }
  __syncthreads();
  __threadfence();                                  // acquire others' stores
}

// ---- preprocess kernels ---------------------------------------------------
__global__ void cvt_bf16_kernel(const float* __restrict s, __bf16* __restrict d, long n) {
  long i = (long)blockIdx.x * blockDim.x + threadIdx.x;
  long st = (long)gridDim.x * blockDim.x;
  for (; i < n; i += st) d[i] = (__bf16)s[i];
}

// split W_mpost/W_lvpost ([256,2048], cols = xt(1024)|h(1024)) into
// Wpx [512,1024] (x-part; mean rows 0..255, lv rows 256..511) and Wph [512,1024]
__global__ void build_post_w_kernel(const float* __restrict Wm, const float* __restrict Wlv,
                                    __bf16* __restrict Wpx, __bf16* __restrict Wph) {
  int i = blockIdx.x * 256 + threadIdx.x;          // 2048 blocks -> 524288
  int r = i >> 10, k = i & 1023;
  const float* W = (r < 256) ? Wm : Wlv;
  int rr = r & 255;
  Wpx[i] = (__bf16)W[rr * 2048 + k];
  Wph[i] = (__bf16)W[rr * 2048 + 1024 + k];
}

__global__ void zero_state_kernel(float* hf, __bf16* hb, __bf16* zb, unsigned* cnt) {
  int i = blockIdx.x * 256 + threadIdx.x;          // 512 blocks -> 131072
  if (i < Bb * DDET) { hf[i] = 0.0f; hb[i] = (__bf16)0.0f; }
  if (i < Bb * DSTO) zb[i] = (__bf16)0.0f;
  if (i == 0) *cnt = 0u;                           // reset scan barrier counter
}

// ---- xproj: for all t, [B,512] = xt @ Wpx^T  (hoisted out of the scan) ----
__global__ void xproj_kernel(const __bf16* __restrict xbf,  // [B,T,DIN]
                             const __bf16* __restrict Wpx,  // [512,1024]
                             float* __restrict xproj) {     // [T,B,512]
  int bid = blockIdx.x;            // T * 8 * 32 = 65536 blocks, 32 threads
  int t  = bid >> 8;
  int mt = (bid >> 5) & 7;
  int nt = bid & 31;
  v8f acc = {};
  const __bf16* A0 = xbf + (size_t)(mt * 16) * (Tt * DIN) + (size_t)t * DIN;
  const __bf16* B0 = Wpx + (size_t)(nt * 16) * 1024;
  for (int k = 0; k < 1024; k += 32) {
    v16bf a = load_frag_nk(A0 + k, Tt * DIN);
    v16bf b = load_frag_nk(B0 + k, 1024);
    acc = wmma_bf16(a, b, acc);
  }
  int col = dcol(), r0 = drow0();
  int n = nt * 16 + col;
  float* dst = xproj + (size_t)t * Bb * 512;
#pragma unroll
  for (int v = 0; v < 8; ++v) {
    int row = mt * 16 + r0 + v;
    dst[(size_t)row * 512 + n] = acc[v];
  }
}

// ---- scan tile bodies -----------------------------------------------------
// One wave computes one 16x16 h_new chunk: gates GEMM + GRU elementwise.
DEVINL void gru_tile(int mt, int nt,
                     const __bf16* __restrict hbf, const float* __restrict hf32,
                     const __bf16* __restrict zbf,
                     const __bf16* __restrict Wih, const __bf16* __restrict Whh,
                     const float* __restrict b_ih, const float* __restrict b_hh,
                     __bf16* __restrict hbf_o, float* __restrict hf32_o,
                     __bf16* __restrict dec_in, int t) {
  v8f accr = {}, accu = {}, accnx = {}, accnh = {};
  {   // z @ W_ih^T (K=256): gates r,u and x-part of n
    const __bf16* A0 = zbf + (size_t)(mt * 16) * DSTO;
    const __bf16* Br = Wih + (size_t)(0 * DDET + nt * 16) * DSTO;
    const __bf16* Bu = Wih + (size_t)(1 * DDET + nt * 16) * DSTO;
    const __bf16* Bn = Wih + (size_t)(2 * DDET + nt * 16) * DSTO;
    for (int k = 0; k < DSTO; k += 32) {
      v16bf a = load_frag_nk(A0 + k, DSTO);
      accr  = wmma_bf16(a, load_frag_nk(Br + k, DSTO), accr);
      accu  = wmma_bf16(a, load_frag_nk(Bu + k, DSTO), accu);
      accnx = wmma_bf16(a, load_frag_nk(Bn + k, DSTO), accnx);
    }
  }
  {   // h @ W_hh^T (K=1024): r,u combined; n-h separate (r gates it)
    const __bf16* A0 = hbf + (size_t)(mt * 16) * DDET;
    const __bf16* Br = Whh + (size_t)(0 * DDET + nt * 16) * DDET;
    const __bf16* Bu = Whh + (size_t)(1 * DDET + nt * 16) * DDET;
    const __bf16* Bn = Whh + (size_t)(2 * DDET + nt * 16) * DDET;
    for (int k = 0; k < DDET; k += 32) {
      v16bf a = load_frag_nk(A0 + k, DDET);
      accr  = wmma_bf16(a, load_frag_nk(Br + k, DDET), accr);
      accu  = wmma_bf16(a, load_frag_nk(Bu + k, DDET), accu);
      accnh = wmma_bf16(a, load_frag_nk(Bn + k, DDET), accnh);
    }
  }
  int col = dcol(), r0 = drow0();
  int n = nt * 16 + col;
  float br  = b_ih[n] + b_hh[n];
  float bu  = b_ih[DDET + n] + b_hh[DDET + n];
  float bnx = b_ih[2 * DDET + n];
  float bnh = b_hh[2 * DDET + n];
#pragma unroll
  for (int v = 0; v < 8; ++v) {
    int row = mt * 16 + r0 + v;                 // batch index b
    float r  = sigmoidf_(accr[v] + br);
    float u  = sigmoidf_(accu[v] + bu);
    float nn = tanhf(accnx[v] + bnx + r * (accnh[v] + bnh));
    float ho = hf32[(size_t)row * DDET + n];
    float hn = (1.0f - u) * nn + u * ho;
    hf32_o[(size_t)row * DDET + n] = hn;
    __bf16 hb = (__bf16)hn;
    hbf_o[(size_t)row * DDET + n] = hb;
    dec_in[(size_t)(row * Tt + t) * DCAT + n] = hb;   // deter part
  }
}

// One wave computes one 16x16 chunk of both posterior heads + reparameterize.
DEVINL void post_tile(int mt, int nt,
                      const __bf16* __restrict hbf, const __bf16* __restrict Wph,
                      const float* __restrict b_m, const float* __restrict b_lv,
                      const float* __restrict xproj_t, const float* __restrict noise_t,
                      __bf16* __restrict zbf_o, __bf16* __restrict dec_in, int t) {
  v8f am = {}, alv = {};
  const __bf16* A0 = hbf + (size_t)(mt * 16) * DDET;
  const __bf16* Bm = Wph + (size_t)(nt * 16) * 1024;
  const __bf16* Bl = Wph + (size_t)(256 + nt * 16) * 1024;
  for (int k = 0; k < DDET; k += 32) {
    v16bf a = load_frag_nk(A0 + k, DDET);
    am  = wmma_bf16(a, load_frag_nk(Bm + k, 1024), am);
    alv = wmma_bf16(a, load_frag_nk(Bl + k, 1024), alv);
  }
  int col = dcol(), r0 = drow0();
  int n = nt * 16 + col;
  float bm = b_m[n], bl = b_lv[n];
#pragma unroll
  for (int v = 0; v < 8; ++v) {
    int row = mt * 16 + r0 + v;
    float mean = am[v]  + bm + xproj_t[(size_t)row * 512 + n];
    float lv   = alv[v] + bl + xproj_t[(size_t)row * 512 + 256 + n];
    float zv   = noise_t[(size_t)row * DSTO + n] * __expf(0.5f * lv) + mean;
    __bf16 zb = (__bf16)zv;
    zbf_o[(size_t)row * DSTO + n] = zb;
    dec_in[(size_t)(row * Tt + t) * DCAT + DDET + n] = zb;  // stoch part
  }
}

// ---- persistent scan: 64 blocks x 8 waves = 512 waves = 512 gate tiles ----
__global__ __launch_bounds__(256) void scan_kernel(
    __bf16* hb0, __bf16* hb1, float* hf0, float* hf1, __bf16* zbf,
    const __bf16* __restrict Wih, const __bf16* __restrict Whh,
    const float* __restrict b_ih, const float* __restrict b_hh,
    const __bf16* __restrict Wph, const float* __restrict b_m,
    const float* __restrict b_lv, const float* __restrict xproj,
    const float* __restrict noise, __bf16* __restrict dec_in, unsigned* cnt) {
  const int w = blockIdx.x * 8 + (threadIdx.x >> 5);   // wave id 0..511 (uniform)
  unsigned bar = 0;
  for (int t = 0; t < Tt; ++t) {
    const bool p = (t & 1) != 0;
    const __bf16* hbi = p ? hb1 : hb0;  __bf16* hbo = p ? hb0 : hb1;
    const float*  hfi = p ? hf1 : hf0;  float*  hfo = p ? hf0 : hf1;
    gru_tile(w >> 6, w & 63, hbi, hfi, zbf, Wih, Whh, b_ih, b_hh,
             hbo, hfo, dec_in, t);
    grid_sync(cnt, ++bar * SCAN_BLOCKS);               // h_new complete
    if (w < 128)
      post_tile(w >> 4, w & 15, hbo, Wph, b_m, b_lv,
                xproj + (size_t)t * Bb * 512, noise + (size_t)t * Bb * DSTO,
                zbf, dec_in, t);
    grid_sync(cnt, ++bar * SCAN_BLOCKS);               // z_new complete
  }
}

// ---- decoder: 4 fused layers, 32-row tile per block, LDS ping-pong --------
DEVINL void mlp_layer32(const __bf16* src, int K, const __bf16* __restrict W,
                        const float* __restrict bias, __bf16* dstLDS,
                        float* __restrict dstG, bool elu) {
  int wv = threadIdx.x >> 5;
  int col = dcol(), r0 = drow0();
  v8f acc[2][8];
#pragma unroll
  for (int m = 0; m < 2; ++m)
#pragma unroll
    for (int i = 0; i < 8; ++i) acc[m][i] = {};
  for (int k = 0; k < K; k += 32) {
    v16bf a0 = load_frag_nk(src + k, K);               // LDS ds_load_b128
    v16bf a1 = load_frag_nk(src + (size_t)16 * K + k, K);
    #pragma unroll
    for (int i = 0; i < 8; ++i) {
      v16bf b = load_frag_nk(W + (size_t)((wv * 8 + i) * 16) * K + k, K);
      acc[0][i] = wmma_bf16(a0, b, acc[0][i]);         // B reused for 2 m-tiles
      acc[1][i] = wmma_bf16(a1, b, acc[1][i]);
    }
  }
#pragma unroll
  for (int m = 0; m < 2; ++m) {
    #pragma unroll
    for (int i = 0; i < 8; ++i) {
      int n = (wv * 8 + i) * 16 + col;
      float bs = bias[n];
      #pragma unroll
      for (int v = 0; v < 8; ++v) {
        float val = acc[m][i][v] + bs;
        if (elu) val = (val > 0.0f) ? val : (__expf(val) - 1.0f);
        int row = m * 16 + r0 + v;
        if (dstLDS) dstLDS[row * 1024 + n] = (__bf16)val;
        else        dstG[(size_t)row * 1024 + n] = val;
      }
    }
  }
}

__global__ __launch_bounds__(256) void decoder_kernel(
    const __bf16* __restrict dec_in,  // [32768,1280]
    const __bf16* __restrict W0, const float* __restrict b0,
    const __bf16* __restrict W1, const float* __restrict b1,
    const __bf16* __restrict W2, const float* __restrict b2,
    const __bf16* __restrict W3, const float* __restrict b3,
    float* __restrict out) {
  extern __shared__ __bf16 smem[];
  __bf16* actA = smem;                 // 32*1280 bf16 (reused with ld=1024)
  __bf16* actB = smem + DRT * DCAT;    // 32*1024 bf16
  size_t rowBase = (size_t)blockIdx.x * DRT;

  // Stage 32 contiguous input rows (81920 B) into LDS via the CDNA5
  // async-to-LDS path: 5120 b128 chunks over 256 lanes x 20 iters.
  {
    const char* g = (const char*)(dec_in + rowBase * DCAT);
    unsigned ldsbase = (unsigned)(unsigned long long)(void*)actA;
#pragma unroll
    for (int i = 0; i < 20; ++i) {
      unsigned off = (unsigned)(i * 256 + threadIdx.x) * 16u;
      unsigned long long ga = (unsigned long long)(g + off);
      unsigned la = ldsbase + off;
      asm volatile("global_load_async_to_lds_b128 %0, %1, off"
                   :: "v"(la), "v"(ga) : "memory");
    }
    asm volatile("s_wait_asynccnt 0x0" ::: "memory");
  }
  __syncthreads();

  mlp_layer32(actA, DCAT, W0, b0, actB, nullptr, true);  __syncthreads();
  mlp_layer32(actB, 1024, W1, b1, actA, nullptr, true);  __syncthreads();
  mlp_layer32(actA, 1024, W2, b2, actB, nullptr, true);  __syncthreads();
  mlp_layer32(actB, 1024, W3, b3, nullptr, out + rowBase * 1024, false);
}

// ---------------------------------------------------------------------------
extern "C" void kernel_launch(void* const* d_in, const int* in_sizes, int n_in,
                              void* d_out, int out_size, void* d_ws, size_t ws_size,
                              hipStream_t stream) {
  (void)in_sizes; (void)n_in; (void)out_size; (void)ws_size;
  const float* x     = (const float*)d_in[0];
  const float* noise = (const float*)d_in[1];
  const float* W_ih  = (const float*)d_in[2];
  const float* W_hh  = (const float*)d_in[3];
  const float* b_ih  = (const float*)d_in[4];
  const float* b_hh  = (const float*)d_in[5];
  const float* W_m   = (const float*)d_in[6];
  const float* b_m   = (const float*)d_in[7];
  const float* W_lv  = (const float*)d_in[8];
  const float* b_lv  = (const float*)d_in[9];
  const float* dW0   = (const float*)d_in[10];
  const float* db0   = (const float*)d_in[11];
  const float* dW1   = (const float*)d_in[12];
  const float* db1   = (const float*)d_in[13];
  const float* dW2   = (const float*)d_in[14];
  const float* db2   = (const float*)d_in[15];
  const float* dW3   = (const float*)d_in[16];
  const float* db3   = (const float*)d_in[17];
  float* out = (float*)d_out;

  char* ws = (char*)d_ws;
  auto carve = [&](size_t bytes) -> char* {
    char* p = ws; ws += (bytes + 255) & ~(size_t)255; return p;
  };
  __bf16* Wih_bf = (__bf16*)carve((size_t)3072 * 256 * 2);
  __bf16* Whh_bf = (__bf16*)carve((size_t)3072 * 1024 * 2);
  __bf16* Wpx_bf = (__bf16*)carve((size_t)512 * 1024 * 2);
  __bf16* Wph_bf = (__bf16*)carve((size_t)512 * 1024 * 2);
  __bf16* W0_bf  = (__bf16*)carve((size_t)1024 * 1280 * 2);
  __bf16* W1_bf  = (__bf16*)carve((size_t)1024 * 1024 * 2);
  __bf16* W2_bf  = (__bf16*)carve((size_t)1024 * 1024 * 2);
  __bf16* W3_bf  = (__bf16*)carve((size_t)1024 * 1024 * 2);
  __bf16* xbf    = (__bf16*)carve((size_t)Bb * Tt * DIN * 2);
  float*  xproj  = (float*) carve((size_t)Tt * Bb * 512 * 4);
  float*  hf[2]  = {(float*)carve((size_t)Bb * DDET * 4),
                    (float*)carve((size_t)Bb * DDET * 4)};
  __bf16* hb[2]  = {(__bf16*)carve((size_t)Bb * DDET * 2),
                    (__bf16*)carve((size_t)Bb * DDET * 2)};
  __bf16* zbf    = (__bf16*)carve((size_t)Bb * DSTO * 2);
  unsigned* cnt  = (unsigned*)carve(256);
  __bf16* dec_in = (__bf16*)carve((size_t)ROWS * DCAT * 2);

  // ---- preprocess: bf16 conversion + state/barrier init ----
  cvt_bf16_kernel<<<512,  256, 0, stream>>>(W_ih, Wih_bf, (long)3072 * 256);
  cvt_bf16_kernel<<<1024, 256, 0, stream>>>(W_hh, Whh_bf, (long)3072 * 1024);
  cvt_bf16_kernel<<<1024, 256, 0, stream>>>(dW0, W0_bf, (long)1024 * 1280);
  cvt_bf16_kernel<<<1024, 256, 0, stream>>>(dW1, W1_bf, (long)1024 * 1024);
  cvt_bf16_kernel<<<1024, 256, 0, stream>>>(dW2, W2_bf, (long)1024 * 1024);
  cvt_bf16_kernel<<<1024, 256, 0, stream>>>(dW3, W3_bf, (long)1024 * 1024);
  cvt_bf16_kernel<<<2048, 256, 0, stream>>>(x, xbf, (long)Bb * Tt * DIN);
  build_post_w_kernel<<<2048, 256, 0, stream>>>(W_m, W_lv, Wpx_bf, Wph_bf);
  zero_state_kernel<<<512, 256, 0, stream>>>(hf[0], hb[0], zbf, cnt);

  // ---- hoisted x-projection for posterior heads ----
  xproj_kernel<<<Tt * 256, 32, 0, stream>>>(xbf, Wpx_bf, xproj);

  // ---- persistent scan: all 256 timesteps in ONE launch ----
  scan_kernel<<<SCAN_BLOCKS, 256, 0, stream>>>(
      hb[0], hb[1], hf[0], hf[1], zbf, Wih_bf, Whh_bf, b_ih, b_hh,
      Wph_bf, b_m, b_lv, xproj, noise, dec_in, cnt);

  // ---- decoder: fused 4-layer MLP over all 32768 rows ----
  int smem_bytes = (DRT * DCAT + DRT * 1024) * 2;   // 147456 B < 320 KB/WGP
  hipFuncSetAttribute((const void*)decoder_kernel,
                      hipFuncAttributeMaxDynamicSharedMemorySize, smem_bytes);
  decoder_kernel<<<ROWS / DRT, 256, smem_bytes, stream>>>(
      dec_in, W0_bf, db0, W1_bf, db1, W2_bf, db2, W3_bf, db3, out);
}
